// Cross_Spatial_Attention_68092411511430
// MI455X (gfx1250) — matrix operand order repaired and checked by reading, not compiled, not run on hardware
//
#include <hip/hip_runtime.h>

// ---------------- problem constants ----------------
#define Bq 2
#define Cc 128
#define Hh 256
#define Ww 256
#define Nn (Hh*Ww)
#define HD 8
#define CH 16
#define C3 384
#define EPSN 1e-12f

typedef __attribute__((ext_vector_type(16))) __bf16 v16bf;
typedef __attribute__((ext_vector_type(8)))  float  v8f;
typedef unsigned int u32x4 __attribute__((ext_vector_type(4)));
typedef int          i32x4 __attribute__((ext_vector_type(4)));
typedef int          i32x8 __attribute__((ext_vector_type(8)));

// ---------------- workspace layout (bytes) ----------------
#define O_TMP    0ull
#define SZ_TMP   ((size_t)Bq*C3*Nn*2)          // qkv pre-dwconv, bf16 (96 MB)
#define O_V      (O_TMP + SZ_TMP)
#define SZ_V     ((size_t)Bq*Cc*Nn*2)          // v post-dwconv, bf16 (32 MB)
#define O_G      (O_V + SZ_V)
#define SZ_G     ((size_t)Bq*HD*CH*CH*4)       // unnormalized Gram QK^T
#define O_QSS    (O_G + SZ_G)
#define SZ_QSS   ((size_t)Bq*Cc*4)
#define O_KSS    (O_QSS + SZ_QSS)
#define O_VSUM   (O_KSS + SZ_QSS)
#define O_ATTNP  (O_VSUM + SZ_QSS)
#define SZ_ATTNP ((size_t)Bq*HD*32*16*2)       // attn packed as WMMA A-fragments
#define O_SPEC   (O_ATTNP + SZ_ATTNP)
#define O_WPACK  (O_SPEC + SZ_QSS)
#define SZ_WPACK ((size_t)24*4*32*16*2)        // qkv_w packed A-fragments bf16
#define O_PPACK  (O_WPACK + SZ_WPACK)
#define SZ_PPACK ((size_t)8*4*32*16*2)         // proj_w packed A-fragments bf16

// ======== Tensor Data Mover: async 2-D f32 tile load, global -> LDS ========
// D# bitfields per CDNA5 ISA.
// Copies tile_d1 rows of tile_d0 contiguous f32 elements; global row stride =
// stride0 elements; rows packed contiguously in LDS starting at lds_off.
__device__ __forceinline__ void tdm_load_2d_f32(unsigned lds_off, const void* gaddr,
                                                unsigned tile_d0, unsigned tile_d1,
                                                unsigned tensor_d0, unsigned tensor_d1,
                                                unsigned long long stride0_elems) {
  unsigned long long ga = (unsigned long long)(uintptr_t)gaddr;
  u32x4 g0;
  g0[0] = 1u;                                            // count=1 (valid user D#)
  g0[1] = lds_off;                                       // lds_addr (bytes)
  g0[2] = (unsigned)(ga & 0xffffffffu);                  // global_addr[31:0]
  g0[3] = (unsigned)((ga >> 32) & 0x01ffffffu)           // global_addr[56:32]
        | (2u << 30);                                    // type = 2 ("image")
  i32x8 g1;
  g1[0] = (int)(2u << 16);                               // data_size=2 (4B); wg_mask=0
  g1[1] = (int)((tensor_d0 & 0xffffu) << 16);            // tensor_dim0[15:0] -> bits 63:48
  g1[2] = (int)(((tensor_d0 >> 16) & 0xffffu)            // tensor_dim0[31:16]
        |        ((tensor_d1 & 0xffffu) << 16));         // tensor_dim1[15:0]
  g1[3] = (int)(((tensor_d1 >> 16) & 0xffffu)            // tensor_dim1[31:16]
        |        ((tile_d0 & 0xffffu) << 16));           // tile_dim0
  g1[4] = (int)(tile_d1 & 0xffffu);                      // tile_dim1; tile_dim2=0
  g1[5] = (int)(unsigned)(stride0_elems & 0xffffffffu);  // tensor_dim0_stride[31:0]
  g1[6] = (int)(unsigned)((stride0_elems >> 32) & 0xffffu); // stride[47:32]; dim1_stride=0
  g1[7] = 0;
  i32x4 gz = {0, 0, 0, 0};
#if __clang_major__ >= 23
  i32x8 gz8 = {0, 0, 0, 0, 0, 0, 0, 0};
  __builtin_amdgcn_tensor_load_to_lds(g0, g1, gz, gz, gz8, 0);
#else
  __builtin_amdgcn_tensor_load_to_lds(g0, g1, gz, gz, 0);
#endif
}

// ================= weight pre-pack into WMMA A-fragment layout =================
// A (16x32 bf16): lane l: m = l&15; kb = l<16?0:16; element e -> K = kb+e
__global__ void k_pack(const float* __restrict__ qkv_w, const float* __restrict__ proj_w,
                       __bf16* __restrict__ wpack, __bf16* __restrict__ ppack) {
  int t = blockIdx.x * blockDim.x + threadIdx.x;   // 0..65535
  int e  = t & 15;
  int l  = (t >> 4) & 31;
  int fk = t >> 9;                                  // fragment id
  int kb = (l < 16) ? 0 : 16;
  int kk = kb + e;
  if (fk < 96) {                                    // qkv_w: 24 M-tiles x 4 K-chunks
    int mt = fk >> 2, kc = fk & 3;
    int oc = mt * 16 + (l & 15);
    int ic = kc * 32 + kk;
    wpack[t] = (__bf16)qkv_w[oc * Cc + ic];
  } else {                                          // proj_w: 8 M-tiles x 4 K-chunks
    int f2 = fk - 96;
    int mt = f2 >> 2, kc = f2 & 3;
    int oc = mt * 16 + (l & 15);
    int ic = kc * 32 + kk;
    ppack[(f2 * 32 + l) * 16 + e] = (__bf16)proj_w[oc * Cc + ic];
  }
}

// ================= K1: qkv 1x1 conv as bf16 WMMA GEMM =================
// tmp[b][384][N] = qkv_w[384][128] @ x[b][128][N]; 128 px per WG, 16 px per wave.
// x tile (128 ch x 128 px f32, 64 KB) staged into LDS by the Tensor Data Mover.
__global__ void k_qkv1x1(const float* __restrict__ x, const __bf16* __restrict__ wpack,
                         __bf16* __restrict__ tmp) {
  __shared__ float sX[128 * 128];       // [ic][px] f32, filled by TDM (LDS offset 0)
  int b  = blockIdx.y;
  int n0 = blockIdx.x * 128;
  int tid = threadIdx.x;
  if (tid < 32) {                       // one wave issues the DMA, waits on TENSORcnt
    tdm_load_2d_f32(0u, x + (size_t)b * Cc * Nn + n0,
                    /*tile_d0=*/128, /*tile_d1=*/Cc,
                    /*tensor_d0=*/Nn, /*tensor_d1=*/Cc,
                    /*stride0=*/(unsigned long long)Nn);
    __builtin_amdgcn_s_wait_tensorcnt(0);
  }
  __syncthreads();
  // Launder the LDS pointer: the TDM descriptor carries the LDS destination as
  // a plain integer, so alias analysis cannot see the write to sX. Without
  // this, the compiler deletes the LDS allocation and folds the reads (seen in
  // round-2 disasm: group_segment_fixed_size 0). The empty asm makes sxp an
  // asm-defined value with a memory clobber, pinning both allocation and data.
  const float* sxp = &sX[0];
  asm volatile("" : "+v"(sxp) : : "memory");

  int wv = tid >> 5, lane = tid & 31;
  int pxb = wv * 16;
  int col = lane & 15;
  int kbB = (lane < 16) ? 0 : 16;
  int rbase = (lane < 16) ? 0 : 8;
  // B fragments built once per wave, reused across all 24 M-tiles
  v16bf bfr[4];
#pragma unroll
  for (int kc = 0; kc < 4; ++kc) {
#pragma unroll
    for (int e = 0; e < 16; ++e)
      bfr[kc][e] = (__bf16)sxp[(kc * 32 + kbB + e) * 128 + pxb + col];
  }
  for (int mt = 0; mt < 24; ++mt) {
    v8f acc = {};
#pragma unroll
    for (int kc = 0; kc < 4; ++kc) {
      v16bf a = *(const v16bf*)(wpack + ((mt * 4 + kc) * 32 + lane) * 16);
      acc = __builtin_amdgcn_wmma_f32_16x16x32_bf16(false, a, false, bfr[kc], (short)0, acc, false, false);
    }
#pragma unroll
    for (int r = 0; r < 8; ++r) {
      int oc = mt * 16 + r + rbase;
      tmp[((size_t)b * C3 + oc) * Nn + n0 + pxb + col] = (__bf16)acc[r];
    }
  }
}

// depthwise 3x3, SAME padding, row-fused
__device__ __forceinline__ float dw3x3(const __bf16* base, const float* w9,
                                       int row, int p, bool up, bool dn) {
  float acc = 0.f;
  const __bf16* rc = base + (size_t)row * Ww;
  if (p > 0)      acc += w9[3] * (float)rc[p - 1];
                  acc += w9[4] * (float)rc[p];
  if (p < Ww - 1) acc += w9[5] * (float)rc[p + 1];
  if (up) {
    const __bf16* r0 = rc - Ww;
    if (p > 0)      acc += w9[0] * (float)r0[p - 1];
                    acc += w9[1] * (float)r0[p];
    if (p < Ww - 1) acc += w9[2] * (float)r0[p + 1];
  }
  if (dn) {
    const __bf16* r2 = rc + Ww;
    if (p > 0)      acc += w9[6] * (float)r2[p - 1];
                    acc += w9[7] * (float)r2[p];
    if (p < Ww - 1) acc += w9[8] * (float)r2[p + 1];
  }
  return acc;
}

// ================= K2: fused dwconv + sa-gate + Gram(QK^T) + v + stats =================
// One WG per (batch,row); wave w == head w. Gram contraction runs over spatial (K=32/WMMA).
__global__ void k_attn_stats(const __bf16* __restrict__ tmp, const float* __restrict__ y,
                             const float* __restrict__ qkv_dw_w,
                             const float* __restrict__ sa_w1, const float* __restrict__ sa_w2,
                             const float* __restrict__ sa_w3,
                             __bf16* __restrict__ vws, float* __restrict__ G,
                             float* __restrict__ qss, float* __restrict__ kss,
                             float* __restrict__ vsum) {
  __shared__ float sW1[16 * 128];
  __shared__ float sW2[16 * 16];
  __shared__ float sW3[16];
  __shared__ float sSa[256];
  int b = blockIdx.y, row = blockIdx.x;
  int n0 = row * Ww;
  int tid = threadIdx.x;
  for (int i = tid; i < 2048; i += 256) sW1[i] = sa_w1[i];
  sW2[tid & 255] = sa_w2[tid & 255];
  if (tid < 16) sW3[tid] = sa_w3[tid];
  __syncthreads();
  // spatial gate sa(y) for this row's 256 pixels (one pixel per thread)
  {
    int p = tid;
    float a1[16];
#pragma unroll
    for (int j = 0; j < 16; ++j) a1[j] = 0.f;
    for (int ic = 0; ic < 128; ++ic) {
      float yv = y[((size_t)b * Cc + ic) * Nn + n0 + p];
#pragma unroll
      for (int j = 0; j < 16; ++j) a1[j] += sW1[j * 128 + ic] * yv;
    }
#pragma unroll
    for (int j = 0; j < 16; ++j) a1[j] = fmaxf(a1[j], 0.f);
    float a2[16];
#pragma unroll
    for (int j = 0; j < 16; ++j) {
      float s = 0.f;
#pragma unroll
      for (int k2 = 0; k2 < 16; ++k2) s += sW2[j * 16 + k2] * a1[k2];
      a2[j] = fmaxf(s, 0.f);
    }
    float s3 = 0.f;
#pragma unroll
    for (int k2 = 0; k2 < 16; ++k2) s3 += sW3[k2] * a2[k2];
    sSa[p] = 1.f / (1.f + expf(-s3));
  }
  __syncthreads();

  int wv = tid >> 5, lane = tid & 31;
  int h = wv;
  bool up = row > 0, dn = row < Hh - 1;
  int cq = h * CH + (lane & 15);        // q channel for this lane (A-fragment row)
  int ck = Cc + h * CH + (lane & 15);   // k channel for this lane (B-fragment column)
  float wq9[9], wk9[9];
#pragma unroll
  for (int i = 0; i < 9; ++i) { wq9[i] = qkv_dw_w[cq * 9 + i]; wk9[i] = qkv_dw_w[ck * 9 + i]; }
  const __bf16* baseq = tmp + ((size_t)b * C3 + cq) * Nn;
  const __bf16* basek = tmp + ((size_t)b * C3 + ck) * Nn;
  // warm L2/L0 for the rows this lane will sweep (global_prefetch_b8)
  __builtin_prefetch(baseq + (size_t)row * Ww, 0, 1);
  __builtin_prefetch(basek + (size_t)row * Ww, 0, 1);
  int kbA = (lane < 16) ? 0 : 16;
  int kbB = (lane < 16) ? 0 : 16;
  float qss_l = 0.f, kss_l = 0.f;
  v8f acc = {};
  for (int g = 0; g < 8; ++g) {         // 8 x 32 spatial = one row
    v16bf af, bfm;
#pragma unroll
    for (int e = 0; e < 16; ++e) {
      int pq = g * 32 + kbA + e;
      float qv = dw3x3(baseq, wq9, row, pq, up, dn) * sSa[pq];
      qss_l += qv * qv;
      af[e] = (__bf16)qv;
      int pk = g * 32 + kbB + e;
      float kv = dw3x3(basek, wk9, row, pk, up, dn);
      kss_l += kv * kv;
      bfm[e] = (__bf16)kv;
    }
    acc = __builtin_amdgcn_wmma_f32_16x16x32_bf16(false, af, false, bfm, (short)0, acc, false, false);
  }
  {
    int rbase = (lane < 16) ? 0 : 8;
    int d = lane & 15;
#pragma unroll
    for (int r = 0; r < 8; ++r)
      atomicAdd(&G[(((size_t)b * HD + h) * CH + (r + rbase)) * CH + d], acc[r]);
  }
  qss_l += __shfl_xor(qss_l, 16, 32);
  kss_l += __shfl_xor(kss_l, 16, 32);
  if (lane < 16) {
    atomicAdd(&qss[(size_t)b * Cc + h * CH + lane], qss_l);
    atomicAdd(&kss[(size_t)b * Cc + h * CH + lane], kss_l);
  }
  // v: dwconv, store bf16, accumulate per-channel sum (for pooled = attn @ mean(v))
  for (int ch = 0; ch < CH; ++ch) {
    int cv = 2 * Cc + h * CH + ch;
    float w9[9];
#pragma unroll
    for (int i = 0; i < 9; ++i) w9[i] = qkv_dw_w[cv * 9 + i];
    const __bf16* basev = tmp + ((size_t)b * C3 + cv) * Nn;
    if (ch + 1 < CH)  // prefetch next channel's center row
      __builtin_prefetch(basev + Nn + (size_t)row * Ww, 0, 1);
    float vacc = 0.f;
    for (int it = 0; it < 8; ++it) {
      int p = it * 32 + lane;
      float vv = dw3x3(basev, w9, row, p, up, dn);
      vws[((size_t)b * Cc + h * CH + ch) * Nn + n0 + p] = (__bf16)vv;
      vacc += vv;
    }
#pragma unroll
    for (int off = 16; off > 0; off >>= 1) vacc += __shfl_xor(vacc, off, 32);
    if (lane == 0) atomicAdd(&vsum[(size_t)b * Cc + h * CH + ch], vacc);
  }
}

// ================= K3: normalize/softmax attn, pack A-frags, spectral gate =================
__global__ void k_finalize(const float* __restrict__ G, const float* __restrict__ qss,
                           const float* __restrict__ kss, const float* __restrict__ vsum,
                           const float* __restrict__ temp,
                           const float* __restrict__ sp_w1, const float* __restrict__ sp_w2,
                           const float* __restrict__ sp_w3,
                           __bf16* __restrict__ attnp, float* __restrict__ spec) {
  __shared__ float sPooled[Bq * Cc];
  __shared__ float sG1[Bq * 16];
  __shared__ float sG2[Bq * 16];
  int tid = threadIdx.x;
  int b = tid >> 7, rem = tid & 127, h = rem >> 4, c = rem & 15;
  {
    float qn = fmaxf(sqrtf(qss[b * Cc + h * CH + c]), EPSN);
    float tt = temp[h];
    float a[16];
    float mx = -1e30f;
#pragma unroll
    for (int d = 0; d < 16; ++d) {
      float kn = fmaxf(sqrtf(kss[b * Cc + h * CH + d]), EPSN);
      float vv = G[(((size_t)b * HD + h) * CH + c) * CH + d] / (qn * kn) * tt;
      a[d] = vv; mx = fmaxf(mx, vv);
    }
    float s = 0.f;
#pragma unroll
    for (int d = 0; d < 16; ++d) { a[d] = expf(a[d] - mx); s += a[d]; }
    float inv = 1.f / s;
#pragma unroll
    for (int d = 0; d < 16; ++d) a[d] *= inv;
    float pl = 0.f;
#pragma unroll
    for (int d = 0; d < 16; ++d) pl += a[d] * vsum[b * Cc + h * CH + d];
    sPooled[b * Cc + h * CH + c] = pl * (1.f / (float)Nn);
    // pack attn row c into WMMA A-fragment layout (K=0..15 valid, 16..31 zero)
#pragma unroll
    for (int half = 0; half < 2; ++half) {
      int l = c + half * 16;
      int kb = (l < 16) ? 0 : 16;
#pragma unroll
      for (int e = 0; e < 16; ++e) {
        int K = kb + e;
        float vv = (K < 16) ? a[K] : 0.f;
        attnp[(((size_t)b * HD + h) * 32 + l) * 16 + e] = (__bf16)vv;
      }
    }
  }
  __syncthreads();
  if (tid < Bq * 16) {
    int bb = tid >> 4, j = tid & 15;
    float s = 0.f;
    for (int ic = 0; ic < 128; ++ic) s += sp_w1[j * 128 + ic] * sPooled[bb * Cc + ic];
    sG1[tid] = 0.5f * s * (1.f + erff(s * 0.70710678118f));
  }
  __syncthreads();
  if (tid < Bq * 16) {
    int bb = tid >> 4, j = tid & 15;
    float s = 0.f;
#pragma unroll
    for (int k2 = 0; k2 < 16; ++k2) s += sp_w2[j * 16 + k2] * sG1[bb * 16 + k2];
    sG2[tid] = 0.5f * s * (1.f + erff(s * 0.70710678118f));
  }
  __syncthreads();
  {
    int bb = tid >> 7, oc = tid & 127;
    float s = 0.f;
#pragma unroll
    for (int k2 = 0; k2 < 16; ++k2) s += sp_w3[oc * 16 + k2] * sG2[bb * 16 + k2];
    spec[bb * Cc + oc] = 1.f / (1.f + expf(-s));
  }
}

// ================= K4: out = attn@v (WMMA) + dwconv(y)*spec, then proj GEMM (WMMA) ======
// One WG per (batch, half-row of 128 px); wave w == head w for phases 1-2.
__global__ void k_out_proj(const __bf16* __restrict__ vws, const float* __restrict__ y,
                           const float* __restrict__ dw_w, const float* __restrict__ spec,
                           const __bf16* __restrict__ attnp, const __bf16* __restrict__ ppack,
                           float* __restrict__ outp) {
  __shared__ __bf16 sXsum[128 * 136];   // [px][c], bf16, pitch 136
  int b   = blockIdx.y;
  int row = blockIdx.x >> 1;
  int px0 = (blockIdx.x & 1) * 128;
  int n0  = row * Ww + px0;
  int tid = threadIdx.x, wv = tid >> 5, lane = tid & 31;
  int h = wv;
  bool up = row > 0, dn = row < Hh - 1;
  // phase 1: y2 = dwconv3x3(y)*spec for this wave's 16 channels
  for (int ch = 0; ch < CH; ++ch) {
    int cg = h * CH + ch;
    float w9[9];
#pragma unroll
    for (int i = 0; i < 9; ++i) w9[i] = dw_w[cg * 9 + i];
    float sp = spec[b * Cc + cg];
    const float* basey = y + ((size_t)b * Cc + cg) * Nn;
    for (int it = 0; it < 4; ++it) {
      int p  = it * 32 + lane;
      int pg = px0 + p;
      float acc = 0.f;
      const float* rc = basey + (size_t)row * Ww;
      if (pg > 0)      acc += w9[3] * rc[pg - 1];
                       acc += w9[4] * rc[pg];
      if (pg < Ww - 1) acc += w9[5] * rc[pg + 1];
      if (up) { const float* r0 = rc - Ww;
        if (pg > 0)      acc += w9[0] * r0[pg - 1];
                         acc += w9[1] * r0[pg];
        if (pg < Ww - 1) acc += w9[2] * r0[pg + 1];
      }
      if (dn) { const float* r2 = rc + Ww;
        if (pg > 0)      acc += w9[6] * r2[pg - 1];
                         acc += w9[7] * r2[pg];
        if (pg < Ww - 1) acc += w9[8] * r2[pg + 1];
      }
      sXsum[p * 136 + cg] = (__bf16)(acc * sp);
    }
  }
  // phase 2: out = attn @ v (contraction K=16, zero-padded to 32), add into sXsum
  {
    v16bf af = *(const v16bf*)(attnp + (((size_t)b * HD + h) * 32 + lane) * 16);
    int colp = lane & 15;
    int kbB  = (lane < 16) ? 0 : 16;
    int rbase = (lane < 16) ? 0 : 8;
    for (int g = 0; g < 8; ++g) {
      v16bf bfm;
#pragma unroll
      for (int e = 0; e < 16; ++e) {
        int K = kbB + e;
        float vv = 0.f;
        if (K < 16)
          vv = (float)vws[((size_t)b * Cc + h * CH + K) * Nn + n0 + g * 16 + colp];
        bfm[e] = (__bf16)vv;
      }
      v8f acc = {};
      acc = __builtin_amdgcn_wmma_f32_16x16x32_bf16(false, af, false, bfm, (short)0, acc, false, false);
#pragma unroll
      for (int r = 0; r < 8; ++r) {
        int c = h * CH + r + rbase;
        int p = g * 16 + colp;
        float old = (float)sXsum[p * 136 + c];
        sXsum[p * 136 + c] = (__bf16)(old + acc[r]);
      }
    }
  }
  __syncthreads();
  // phase 3: final = proj_w @ (out + y2)
  {
    int pxb = wv * 16;
    int colp = lane & 15;
    int kbB  = (lane < 16) ? 0 : 16;
    int rbase = (lane < 16) ? 0 : 8;
    for (int mt = 0; mt < 8; ++mt) {
      v8f acc = {};
#pragma unroll
      for (int kc = 0; kc < 4; ++kc) {
        v16bf a  = *(const v16bf*)(ppack + ((mt * 4 + kc) * 32 + lane) * 16);
        v16bf bm = *(const v16bf*)(&sXsum[(pxb + colp) * 136 + kc * 32 + kbB]);
        acc = __builtin_amdgcn_wmma_f32_16x16x32_bf16(false, a, false, bm, (short)0, acc, false, false);
      }
#pragma unroll
      for (int r = 0; r < 8; ++r) {
        int oc = mt * 16 + r + rbase;
        outp[((size_t)b * Cc + oc) * Nn + n0 + pxb + colp] = acc[r];
      }
    }
  }
}

// ================= host launcher =================
extern "C" void kernel_launch(void* const* d_in, const int* in_sizes, int n_in,
                              void* d_out, int out_size, void* d_ws, size_t ws_size,
                              hipStream_t stream) {
  (void)in_sizes; (void)n_in; (void)out_size; (void)ws_size;
  const float* x        = (const float*)d_in[0];
  const float* y        = (const float*)d_in[1];
  const float* qkv_w    = (const float*)d_in[2];
  const float* qkv_dw_w = (const float*)d_in[3];
  const float* proj_w   = (const float*)d_in[4];
  const float* sa_w1    = (const float*)d_in[5];
  const float* sa_w2    = (const float*)d_in[6];
  const float* sa_w3    = (const float*)d_in[7];
  const float* sp_w1    = (const float*)d_in[8];
  const float* sp_w2    = (const float*)d_in[9];
  const float* sp_w3    = (const float*)d_in[10];
  const float* dw_w     = (const float*)d_in[11];
  const float* temp     = (const float*)d_in[12];

  char* ws = (char*)d_ws;
  __bf16* tmp   = (__bf16*)(ws + O_TMP);
  __bf16* vws   = (__bf16*)(ws + O_V);
  float*  G     = (float*)(ws + O_G);
  float*  qss   = (float*)(ws + O_QSS);
  float*  kss   = (float*)(ws + O_KSS);
  float*  vsum  = (float*)(ws + O_VSUM);
  __bf16* attnp = (__bf16*)(ws + O_ATTNP);
  float*  spec  = (float*)(ws + O_SPEC);
  __bf16* wpack = (__bf16*)(ws + O_WPACK);
  __bf16* ppack = (__bf16*)(ws + O_PPACK);

  // zero the atomic accumulators (G, qss, kss, vsum) every call — graph-capture safe
  hipMemsetAsync(ws + O_G, 0, (size_t)(SZ_G + 3 * SZ_QSS), stream);

  k_pack<<<256, 256, 0, stream>>>(qkv_w, proj_w, wpack, ppack);
  k_qkv1x1<<<dim3(Nn / 128, Bq), 256, 0, stream>>>(x, wpack, tmp);
  k_attn_stats<<<dim3(Hh, Bq), 256, 0, stream>>>(tmp, y, qkv_dw_w, sa_w1, sa_w2, sa_w3,
                                                 vws, G, qss, kss, vsum);
  k_finalize<<<1, 256, 0, stream>>>(G, qss, kss, vsum, temp, sp_w1, sp_w2, sp_w3, attnp, spec);
  k_out_proj<<<dim3(Hh * 2, Bq), 256, 0, stream>>>(vws, y, dw_w, spec, attnp, ppack,
                                                   (float*)d_out);
}
